// rnn_model_89283780149393
// MI455X (gfx1250) — matrix operand order repaired and checked by reading, not compiled, not run on hardware
//
#include <hip/hip_runtime.h>
#include <hip/hip_bf16.h>
#include <math.h>

// ---------------------------------------------------------------------------
// Elman RNN LM on MI455X (gfx1250), bf16 WMMA path.
//   S=2048, B=64, V=32000, H=1024
// d_out: [64,2] float32.
// ---------------------------------------------------------------------------

typedef __attribute__((ext_vector_type(16))) __bf16 v16bf;
typedef __attribute__((ext_vector_type(8)))  __bf16 v8bf;
typedef __attribute__((ext_vector_type(8)))  float  v8f;
typedef __attribute__((ext_vector_type(4)))  float  v4f;

#define Hdim 1024
#define Bdim 64
#define Sdim 2048
#define KCHUNKS 32                 // 1024 / 32
#define PACK_ELEMS (64*32*32*16)   // [n_tile][k_chunk][lane][j]

// --- B-operand pack -----------------------------------------------------------
// WMMA 16x16x32 bf16 B layout (32x16, KxN): lane<16 -> N=lane, elem j<8 -> K=j, j>=8 -> K=16+(j-8);
// lane>=16 -> N=lane-16, same but K += 8.  We pack W (row-major [n][k]) so each
// lane reads its 16 bf16 (32 bytes, contiguous) per (n_tile, k_chunk).
// Also zero-initialize h0.
__global__ __launch_bounds__(256)
void pack_kernel(const float* __restrict__ Wih, const float* __restrict__ Whh,
                 __bf16* __restrict__ Bih, __bf16* __restrict__ Bhh,
                 __bf16* __restrict__ h0) {
  int idx = blockIdx.x * blockDim.x + threadIdx.x;
  if (idx < 2 * PACK_ELEMS) {
    const float* W = (idx < PACK_ELEMS) ? Wih : Whh;
    __bf16* Bp     = (idx < PACK_ELEMS) ? Bih : Bhh;
    int p    = (idx < PACK_ELEMS) ? idx : idx - PACK_ELEMS;
    int j    = p & 15;
    int lane = (p >> 4) & 31;
    int kc   = (p >> 9) & 31;
    int nt   = p >> 14;
    int n = nt * 16 + (lane & 15);
    int k = kc * 32 + ((j < 8) ? j : j + 8) + ((lane >= 16) ? 8 : 0);
    Bp[p] = (__bf16)W[n * Hdim + k];
  } else if (idx < 2 * PACK_ELEMS + Bdim * Hdim) {
    h0[idx - 2 * PACK_ELEMS] = (__bf16)0.0f;
  }
}

// --- A-operand helper ---------------------------------------------------------
// WMMA 16x16x32 bf16 A layout (16x32, MxK): lane<16 -> M=lane,
//   elems j=0..7 -> K=j,  j=8..15 -> K=16+(j-8);
// lane>=16 -> M=lane-16, same but K += 8.
// With row-major storage that is two contiguous 8-element runs.
__device__ __forceinline__ v16bf make_a_bf16(const __bf16* rowptr, int kb, int off) {
  v8bf lo = *(const v8bf*)(rowptr + kb + off);
  v8bf hi = *(const v8bf*)(rowptr + kb + 16 + off);
  v16bf a;
#pragma unroll
  for (int j = 0; j < 8; ++j) { a[j] = lo[j]; a[8 + j] = hi[j]; }
  return a;
}

// --- Kernel 1: xp = embed-gather GEMM -----------------------------------------
// xp[m][n] = sum_k emb[tok[m]][k] * W_ih[n][k] + b_ih[n], m in [0, S*B)
// One wave = 1 M-tile x 8 N-tiles (A reused 8x -> ~4.3 GB A traffic total,
// and 8 independent WMMA chains per A-load to hide matrix-pipe latency).
__global__ __launch_bounds__(256)
void xp_gemm_kernel(const int* __restrict__ tok, const float* __restrict__ emb,
                    const __bf16* __restrict__ Bih, const float* __restrict__ bih,
                    __bf16* __restrict__ xp) {
  int wg   = (blockIdx.x * blockDim.x + threadIdx.x) >> 5; // 0..65535
  int lane = threadIdx.x & 31;
  int mt = wg >> 3;   // 0..8191 (M tile of 16 rows)
  int ng = wg & 7;    // group of 8 N tiles
  int row = mt * 16 + (lane & 15);
  const float* arow = emb + (size_t)tok[row] * Hdim;
  int off = (lane >= 16) ? 8 : 0;

  v8f acc[8] = {v8f{}, v8f{}, v8f{}, v8f{}, v8f{}, v8f{}, v8f{}, v8f{}};
  for (int kc = 0; kc < KCHUNKS; ++kc) {
    int kb = kc * 32;
    // A: 16 f32 in two contiguous 32B runs, convert to bf16 (RNE)
    v4f f0 = *(const v4f*)(arow + kb + off);
    v4f f1 = *(const v4f*)(arow + kb + off + 4);
    v4f f2 = *(const v4f*)(arow + kb + 16 + off);
    v4f f3 = *(const v4f*)(arow + kb + 16 + off + 4);
    float af[16];
    *(v4f*)(af + 0) = f0; *(v4f*)(af + 4)  = f1;
    *(v4f*)(af + 8) = f2; *(v4f*)(af + 12) = f3;
    v16bf a;
#pragma unroll
    for (int j = 0; j < 16; ++j) a[j] = (__bf16)af[j];
#pragma unroll
    for (int nt = 0; nt < 8; ++nt) {
      const v16bf b = *(const v16bf*)(Bih +
          ((((size_t)(ng * 8 + nt) * KCHUNKS) + kc) * 32 + lane) * 16);
      acc[nt] = __builtin_amdgcn_wmma_f32_16x16x32_bf16(
          false, a, false, b, (short)0, acc[nt], false, false);
    }
  }
  // C layout: VGPR r -> M = r + (lane>=16 ? 8 : 0), N = lane&15
  int mro = (lane >= 16) ? 8 : 0;
#pragma unroll
  for (int nt = 0; nt < 8; ++nt) {
    int n = (ng * 8 + nt) * 16 + (lane & 15);
    float bias = bih[n];
#pragma unroll
    for (int r = 0; r < 8; ++r) {
      int m = mt * 16 + r + mro;
      xp[(size_t)m * Hdim + n] = (__bf16)(acc[nt][r] + bias);
    }
  }
}

// --- Kernel 2: one recurrence step --------------------------------------------
// hnext = tanh(xp_t + hprev @ W_hh^T + b_hh), hprev/hnext row-major bf16 [64][1024]
// 128 waves (16 blocks x 8) = 64 N-tiles x 2 M-groups; each wave owns 2 M-tiles
// (2 independent WMMA chains, W_hh pack read 2x/step instead of 4x).
__global__ __launch_bounds__(256)
void rnn_step_kernel(const __bf16* __restrict__ xp_t, const __bf16* __restrict__ Bhh,
                     const float* __restrict__ bhh, const __bf16* __restrict__ hprev,
                     __bf16* __restrict__ hnext) {
  int wg   = (blockIdx.x * blockDim.x + threadIdx.x) >> 5; // 0..127
  int lane = threadIdx.x & 31;
  int nt = wg & 63;   // N tile 0..63
  int mg = wg >> 6;   // M group 0..1 -> M tiles {2mg, 2mg+1}
  int off = (lane >= 16) ? 8 : 0;
  const __bf16* arow0 = hprev + (size_t)((mg * 2 + 0) * 16 + (lane & 15)) * Hdim;
  const __bf16* arow1 = hprev + (size_t)((mg * 2 + 1) * 16 + (lane & 15)) * Hdim;

  v8f acc0 = {}, acc1 = {};
  for (int kc = 0; kc < KCHUNKS; ++kc) {
    int kb = kc * 32;
    v16bf a0 = make_a_bf16(arow0, kb, off);
    v16bf a1 = make_a_bf16(arow1, kb, off);
    const v16bf b = *(const v16bf*)(Bhh +
        (((size_t)nt * KCHUNKS + kc) * 32 + lane) * 16);
    acc0 = __builtin_amdgcn_wmma_f32_16x16x32_bf16(
        false, a0, false, b, (short)0, acc0, false, false);
    acc1 = __builtin_amdgcn_wmma_f32_16x16x32_bf16(
        false, a1, false, b, (short)0, acc1, false, false);
  }
  int mro = (lane >= 16) ? 8 : 0;
  int n = nt * 16 + (lane & 15);
  float bias = bhh[n];
#pragma unroll
  for (int r = 0; r < 8; ++r) {
    int m0 = (mg * 2 + 0) * 16 + r + mro;
    int m1 = (mg * 2 + 1) * 16 + r + mro;
    float v0 = acc0[r] + (float)xp_t[(size_t)m0 * Hdim + n] + bias;
    float v1 = acc1[r] + (float)xp_t[(size_t)m1 * Hdim + n] + bias;
    hnext[(size_t)m0 * Hdim + n] = (__bf16)tanhf(v0);
    hnext[(size_t)m1 * Hdim + n] = (__bf16)tanhf(v1);
  }
}

// --- Kernel 3: classifier -----------------------------------------------------
__global__ __launch_bounds__(128)
void fc_kernel(const __bf16* __restrict__ h, const float* __restrict__ Wfc,
               const float* __restrict__ bfc, float* __restrict__ out) {
  int tid = threadIdx.x;        // 0..127
  int b = tid >> 1, c = tid & 1;
  float s = bfc[c];
  for (int k = 0; k < Hdim; ++k)
    s += (float)h[(size_t)b * Hdim + k] * Wfc[c * Hdim + k];
  out[b * 2 + c] = s;
}

// ---------------------------------------------------------------------------
extern "C" void kernel_launch(void* const* d_in, const int* in_sizes, int n_in,
                              void* d_out, int out_size, void* d_ws, size_t ws_size,
                              hipStream_t stream) {
  const int*   tok = (const int*)d_in[0];   // [S,B] tokens
  const float* emb = (const float*)d_in[1]; // [V,H]
  const float* Wih = (const float*)d_in[2]; // [H,H]
  const float* Whh = (const float*)d_in[3]; // [H,H]
  const float* bih = (const float*)d_in[4]; // [H]
  const float* bhh = (const float*)d_in[5]; // [H]
  const float* Wfc = (const float*)d_in[6]; // [2,H]
  const float* bfc = (const float*)d_in[7]; // [2]
  float* out = (float*)d_out;

  // Workspace carve-up (~260.4 MB total)
  char* ws = (char*)d_ws;
  __bf16* xp = (__bf16*)ws;                                   // 256 MB
  size_t xp_bytes = (size_t)Sdim * Bdim * Hdim * sizeof(__bf16);
  __bf16* Bih = (__bf16*)(ws + xp_bytes);                     // 2 MB
  __bf16* Bhh = Bih + PACK_ELEMS;                             // 2 MB
  __bf16* h0  = Bhh + PACK_ELEMS;                             // 128 KB
  __bf16* h1  = h0 + Bdim * Hdim;                             // 128 KB

  // 1) pack weights into WMMA B-operand layout, zero h0
  int packTotal = 2 * PACK_ELEMS + Bdim * Hdim;
  pack_kernel<<<(packTotal + 255) / 256, 256, 0, stream>>>(Wih, Whh, Bih, Bhh, h0);

  // 2) xp = gather+GEMM (65536 waves, 8 N-tiles per wave)
  xp_gemm_kernel<<<8192, 256, 0, stream>>>(tok, emb, Bih, bih, xp);

  // 3) sequential scan: 2048 step kernels (graph replay amortizes launches)
  __bf16* hp = h0; __bf16* hn = h1;
  for (int t = 0; t < Sdim; ++t) {
    rnn_step_kernel<<<16, 256, 0, stream>>>(xp + (size_t)t * Bdim * Hdim,
                                            Bhh, bhh, hp, hn);
    __bf16* tmp = hp; hp = hn; hn = tmp;
  }

  // 4) classifier on final hidden state (in hp after even # of swaps)
  fc_kernel<<<1, 128, 0, stream>>>(hp, Wfc, bfc, out);
}
